// KDLoss_19112604467964
// MI455X (gfx1250) — matrix operand order repaired and checked by reading, not compiled, not run on hardware
//
#include <hip/hip_runtime.h>
#include <math.h>

// ---------------------------------------------------------------------------
// KD loss: sq = sum((t@t.T - r@r.T)^2), out = sqrt(sq*N + 1e-5)
// t = l2norm(targets), r = l2norm(results); N=4096, D=2048 f32 in, f32 scalar out.
//
// MI455X v2:
//  - normalize -> bf16 (32MB total, L2-resident on 192MB L2)
//  - 256-thread (8-wave) blocks, 128x64 Gram-diff tiles, symmetry-halved grid
//  - B tiles double-buffered in LDS via GLOBAL_LOAD_ASYNC_TO_LDS_B128
//    (ASYNCcnt + s_wait_asynccnt), A fragments as direct global_load_b128
//  - v_wmma_f32_16x16x32_bf16, fused weighted diff^2 reduce, scalar atomicAdd
// ---------------------------------------------------------------------------

#define NROWS 4096
#define DCOLS 2048
#define BPAD  40   // padded K-stride (elements) for LDS B tile: 32 + 8

typedef __attribute__((ext_vector_type(16))) __bf16 v16bf;
typedef __attribute__((ext_vector_type(8)))  float  v8f;

union FragAB {
    v16bf v;
    uint4 q[2];
};

// --------------------------- row-normalize to bf16 -------------------------
__global__ void __launch_bounds__(256)
normalize_bf16_kernel(const float* __restrict__ targets,
                      const float* __restrict__ results,
                      __bf16* __restrict__ tb,
                      __bf16* __restrict__ rb) {
    const int row = blockIdx.x;
    const float* src = (blockIdx.y == 0) ? targets : results;
    __bf16*      dst = (blockIdx.y == 0) ? tb : rb;

    const float* x = src + (size_t)row * DCOLS;
    float vals[DCOLS / 256];
    float ss = 0.0f;
#pragma unroll
    for (int i = 0; i < DCOLS / 256; ++i) {
        float v = x[threadIdx.x + i * 256];
        vals[i] = v;
        ss += v * v;
    }
#pragma unroll
    for (int off = 16; off > 0; off >>= 1)
        ss += __shfl_down(ss, off, 32);

    __shared__ float wsum[8];
    if ((threadIdx.x & 31) == 0) wsum[threadIdx.x >> 5] = ss;
    __syncthreads();
    float tot = 0.0f;
#pragma unroll
    for (int w = 0; w < 8; ++w) tot += wsum[w];

    const float scale = 1.0f / fmaxf(sqrtf(tot), 1e-12f);
    __bf16* y = dst + (size_t)row * DCOLS;
#pragma unroll
    for (int i = 0; i < DCOLS / 256; ++i)
        y[threadIdx.x + i * 256] = (__bf16)(vals[i] * scale);
}

// ----------------- async B-tile stage: global -> LDS (CDNA5) ---------------
// Stage 64 cols x 32 K of both matrices (8KB) into one LDS buffer.
// 512 x 16B chunks over 256 threads = 2 async b128 loads per thread.
__device__ __forceinline__ void issue_b_loads(const __bf16* __restrict__ tb,
                                              const __bf16* __restrict__ rb,
                                              __bf16* lds_t, __bf16* lds_r,
                                              int j0, int kb, int tid) {
#pragma unroll
    for (int s = 0; s < 2; ++s) {
        const int c     = tid * 2 + s;      // 0..511
        const int mat   = c >> 8;           // 0: t, 1: r
        const int col   = (c & 255) >> 2;   // 0..63
        const int piece = c & 3;            // 4 x 8 bf16 per column
        const __bf16* g = (mat ? rb : tb) +
                          (size_t)(j0 + col) * DCOLS + kb + piece * 8;
        __bf16* l = (mat ? lds_r : lds_t) + col * BPAD + piece * 8;
        const uint32_t lds_off = (uint32_t)(uintptr_t)l;  // LDS aperture: addr[31:0]
        asm volatile("global_load_async_to_lds_b128 %0, %1, off"
                     :: "v"(lds_off), "v"(g) : "memory");
    }
}

// ------------------------- fused Gram-diff reduction -----------------------
// grid (64, 32): 64-col x 128-row tiles; keep only tiles with any j >= i.
// Wave w owns rows i0 = by*128 + w*16; per K-step: 8 WMMAs (4 col-blocks x 2).
__global__ void __launch_bounds__(256)
gram_diff_kernel(const __bf16* __restrict__ tb,
                 const __bf16* __restrict__ rb,
                 float* __restrict__ acc) {
    if (blockIdx.x < 2 * blockIdx.y) return;   // tile fully below diagonal

    const int tid  = threadIdx.x;
    const int lane = tid & 31;
    const int wave = tid >> 5;
    const int i0 = blockIdx.y * 128 + wave * 16;
    const int j0 = blockIdx.x * 64;

    const int m  = lane & 15;
    const int hi = lane >> 4;

    // [buf][mat][col*BPAD + k]
    __shared__ __bf16 ldsB[2][2][64 * BPAD];

    v8f accT[4], accR[4];
#pragma unroll
    for (int cb = 0; cb < 4; ++cb) {
        accT[cb] = (v8f){0.f, 0.f, 0.f, 0.f, 0.f, 0.f, 0.f, 0.f};
        accR[cb] = (v8f){0.f, 0.f, 0.f, 0.f, 0.f, 0.f, 0.f, 0.f};
    }

    const __bf16* arow_t = tb + (size_t)(i0 + m) * DCOLS;
    const __bf16* arow_r = rb + (size_t)(i0 + m) * DCOLS;

    issue_b_loads(tb, rb, &ldsB[0][0][0], &ldsB[0][1][0], j0, 0, tid);

    const int NIT = DCOLS / 32;
    for (int it = 0; it < NIT; ++it) {
        const int kb  = it * 32;
        const int cur = it & 1;
        if (it + 1 < NIT) {
            issue_b_loads(tb, rb, &ldsB[cur ^ 1][0][0], &ldsB[cur ^ 1][1][0],
                          j0, kb + 32, tid);
            asm volatile("s_wait_asynccnt 0x2" ::: "memory");
        } else {
            asm volatile("s_wait_asynccnt 0x0" ::: "memory");
        }
        __syncthreads();

        // A fragments (16x32 bf16): K = hi*16 + e, e=0..15 contiguous
        FragAB aT, aR;
        aT.q[0] = *(const uint4*)(arow_t + kb + hi * 16);
        aT.q[1] = *(const uint4*)(arow_t + kb + hi * 16 + 8);
        aR.q[0] = *(const uint4*)(arow_r + kb + hi * 16);
        aR.q[1] = *(const uint4*)(arow_r + kb + hi * 16 + 8);

#pragma unroll
        for (int cb = 0; cb < 4; ++cb) {
            // B fragments (32x16 bf16): lane n, K = hi*16 + e, e=0..15 contiguous
            const __bf16* lt = &ldsB[cur][0][(cb * 16 + m) * BPAD + hi * 16];
            const __bf16* lr = &ldsB[cur][1][(cb * 16 + m) * BPAD + hi * 16];
            FragAB bT, bR;
            bT.q[0] = *(const uint4*)(lt);
            bT.q[1] = *(const uint4*)(lt + 8);
            bR.q[0] = *(const uint4*)(lr);
            bR.q[1] = *(const uint4*)(lr + 8);
            accT[cb] = __builtin_amdgcn_wmma_f32_16x16x32_bf16(
                false, aT.v, false, bT.v, (short)0, accT[cb], false, false);
            accR[cb] = __builtin_amdgcn_wmma_f32_16x16x32_bf16(
                false, aR.v, false, bR.v, (short)0, accR[cb], false, false);
        }
        __syncthreads();   // protect buf `cur` before it is re-filled
    }

    // weighted (j>i: x2, j==i: x1, j<i: x0) diff^2 reduction.
    // C/D layout: lane L -> col n = L&15; VGPR v -> row = v + (L>=16 ? 8 : 0)
    float local = 0.0f;
#pragma unroll
    for (int cb = 0; cb < 4; ++cb) {
#pragma unroll
        for (int v = 0; v < 8; ++v) {
            const int i = i0 + hi * 8 + v;
            const int j = j0 + cb * 16 + m;
            const float w = (j > i) ? 2.0f : ((j == i) ? 1.0f : 0.0f);
            const float d = accT[cb][v] - accR[cb][v];
            local += w * d * d;
        }
    }
#pragma unroll
    for (int off = 16; off > 0; off >>= 1)
        local += __shfl_down(local, off, 32);
    if (lane == 0) atomicAdd(acc, local);
}

// ------------------------------- finalize ----------------------------------
__global__ void finalize_kernel(const float* __restrict__ acc,
                                float* __restrict__ out) {
    out[0] = sqrtf(acc[0] * (float)NROWS + 1e-5f);
}

// ------------------------------- launcher ----------------------------------
extern "C" void kernel_launch(void* const* d_in, const int* in_sizes, int n_in,
                              void* d_out, int out_size, void* d_ws, size_t ws_size,
                              hipStream_t stream) {
    const float* results = (const float*)d_in[0];
    const float* targets = (const float*)d_in[1];

    char* ws = (char*)d_ws;
    __bf16* tb  = (__bf16*)ws;                                  // 16 MB
    __bf16* rb  = (__bf16*)(ws + (size_t)NROWS * DCOLS * 2);    // 16 MB
    float*  acc = (float*)(ws + (size_t)NROWS * DCOLS * 4);     // 4 B

    hipMemsetAsync(acc, 0, sizeof(float), stream);

    dim3 gN(NROWS, 2);
    normalize_bf16_kernel<<<gN, 256, 0, stream>>>(targets, results, tb, rb);

    dim3 gG(NROWS / 64, NROWS / 128);
    gram_diff_kernel<<<gG, 256, 0, stream>>>(tb, rb, acc);

    finalize_kernel<<<1, 1, 0, stream>>>(acc, (float*)d_out);
}
